// IPNN_5944234738098
// MI455X (gfx1250) — matrix operand order repaired and checked by reading, not compiled, not run on hardware
//
#include <hip/hip_runtime.h>
#include <hip/hip_bf16.h>
#include <math.h>
#include <stdint.h>

// ---------------- CDNA5 (gfx1250, wave32) WMMA types ----------------
typedef __attribute__((ext_vector_type(16))) __bf16 v16bf;
typedef __attribute__((ext_vector_type(8)))  __bf16 v8bf;
typedef __attribute__((ext_vector_type(8)))  float  v8f;

// Problem constants (from reference)
#define BATCH      16384
#define CONT_F     13
#define CATE_F     26
#define NFIELDS    39
#define EMBD       64
#define NCOMB      741            // 39*38/2
#define FIRST_RAW  3237           // 39*64 + 741
#define KPAD1      3264           // FIRST_RAW rounded up to 32
#define HID        1024

// GEMM tiling
#define BM 128
#define BN 128
#define BK 32

// Wait for all outstanding async global->LDS transfers (ASYNCcnt == 0).
#define WAIT_ASYNC() asm volatile("s_wait_asynccnt 0x0" ::: "memory")

// ------------------------------------------------------------------
// Transpose fp32 [K][N] -> bf16 [N][Kp], zero-padding K..Kp-1.
// ------------------------------------------------------------------
__global__ __launch_bounds__(256)
void transpose_to_bf16(const float* __restrict__ W, __bf16* __restrict__ Wt,
                       int K, int N, int Kp) {
    size_t i = (size_t)blockIdx.x * 256 + threadIdx.x;
    size_t total = (size_t)N * (size_t)Kp;
    if (i >= total) return;
    int n = (int)(i / Kp);
    int k = (int)(i % Kp);
    Wt[i] = (k < K) ? (__bf16)W[(size_t)k * N + n] : (__bf16)0.0f;
}

// ------------------------------------------------------------------
// Build X (bf16, [B][KPAD1]):
//   X[:, f*64+d]     = cc[f][d]   (cont: emb[f]*cont, cate: gathered emb)
//   X[:, 2496 + p]   = <cc[f], cc[g]>  for triu pair p (row-major order)
//   X[:, 3237:3264]  = 0
// One workgroup (256 threads) per sample; cc staged in LDS.
// ------------------------------------------------------------------
__global__ __launch_bounds__(256)
void build_X(const float* __restrict__ conts, const int* __restrict__ cates,
             const float* __restrict__ emb, __bf16* __restrict__ X) {
    const int b   = blockIdx.x;
    const int tid = threadIdx.x;
    __shared__ float cc[NFIELDS][EMBD + 1];   // +1 pad: kills LDS bank conflicts

    __bf16* xrow = X + (size_t)b * KPAD1;

    // Fill cc and the first 2496 entries of X
    for (int idx = tid; idx < NFIELDS * EMBD; idx += 256) {
        const int f = idx >> 6;
        const int d = idx & 63;
        float val;
        if (f < CONT_F) {
            val = emb[f * EMBD + d] * conts[(size_t)b * CONT_F + f];
        } else {
            const int r = cates[(size_t)b * CATE_F + (f - CONT_F)];
            val = emb[(size_t)r * EMBD + d];
        }
        cc[f][d] = val;
        xrow[idx] = (__bf16)val;
    }
    // Zero pad tail (K padding for the WMMA GEMM)
    for (int idx = tid; idx < (KPAD1 - FIRST_RAW); idx += 256)
        xrow[FIRST_RAW + idx] = (__bf16)0.0f;

    __syncthreads();

    // 741 pairwise inner products from LDS
    for (int p = tid; p < NCOMB; p += 256) {
        int f = 0, base = 0;
        while (p >= base + (NFIELDS - 1 - f)) { base += NFIELDS - 1 - f; ++f; }
        const int g = f + 1 + (p - base);
        float s = 0.0f;
        #pragma unroll
        for (int d = 0; d < EMBD; ++d)
            s = fmaf(cc[f][d], cc[g][d], s);
        xrow[NFIELDS * EMBD + p] = (__bf16)s;
    }
}

// ------------------------------------------------------------------
// C[M][N] (fp32) = A[M][K] (bf16, row-major) x Bt[N][K] (bf16, row-major
// = B transposed). M,N multiples of 128, K multiple of 32.
// 8 waves/WG (wave32): 4 wave-rows x 2 wave-cols, each wave 32x64 =
// 2x4 subtiles of 16x16, one v_wmma_f32_16x16x32_bf16 per subtile per
// BK chunk. Double-buffered LDS, fed by GLOBAL_LOAD_ASYNC_TO_LDS_B128
// (ASYNCcnt path — no VGPR round-trip for tile data).
// Fragments use contiguous k = hi*16 + e in both A 16x32 and B 32x16 (not the ISA table order).
// ------------------------------------------------------------------
__global__ __launch_bounds__(256)
void gemm_bf16_wmma(const __bf16* __restrict__ A, const __bf16* __restrict__ Bt,
                    float* __restrict__ C, int M, int N, int K) {
    __shared__ __bf16 sA[2][BM][BK];
    __shared__ __bf16 sB[2][BN][BK];

    const int tid  = threadIdx.x;
    const int wid  = tid >> 5;
    const int lane = tid & 31;
    const int hi   = lane >> 4;       // lane group (0: lanes 0-15, 1: 16-31)
    const int lm   = lane & 15;

    const int m0 = blockIdx.y * BM;
    const int n0 = blockIdx.x * BN;

    const int wr = wid >> 1;          // 0..3  -> 32-row slab
    const int wc = wid & 1;           // 0..1  -> 64-col slab
    const int wm = wr * 32;
    const int wn = wc * 64;

    v8f acc[2][4];
    #pragma unroll
    for (int i = 0; i < 2; ++i)
        #pragma unroll
        for (int j = 0; j < 4; ++j)
            #pragma unroll
            for (int e = 0; e < 8; ++e) acc[i][j][e] = 0.0f;

    // Global->LDS (async): each thread moves 32B of A-tile and 32B of B-tile.
    const int lr = tid >> 1;          // 0..127: tile row
    const int lc = (tid & 1) * 16;    // 0 or 16: bf16 column within BK

    auto load_tile_async = [&](int buf, int kk) {
        const __bf16* ap = A  + (size_t)(m0 + lr) * K + kk + lc;
        const __bf16* bp = Bt + (size_t)(n0 + lr) * K + kk + lc;
        // LDS aperture keeps the byte offset in addr[31:0] (ISA 10.2) —
        // low 32 bits of the generic pointer are the DS address.
        unsigned la = (unsigned)(uintptr_t)(void*)&sA[buf][lr][lc];
        unsigned lb = (unsigned)(uintptr_t)(void*)&sB[buf][lr][lc];
        asm volatile(
            "global_load_async_to_lds_b128 %0, %1, off\n\t"
            "global_load_async_to_lds_b128 %0, %1, off offset:16"
            :: "v"(la), "v"(ap) : "memory");
        asm volatile(
            "global_load_async_to_lds_b128 %0, %1, off\n\t"
            "global_load_async_to_lds_b128 %0, %1, off offset:16"
            :: "v"(lb), "v"(bp) : "memory");
    };

    const int nk = K / BK;
    load_tile_async(0, 0);
    WAIT_ASYNC();
    __syncthreads();

    for (int kc = 0; kc < nk; ++kc) {
        const int cur = kc & 1;
        if (kc + 1 < nk) load_tile_async(cur ^ 1, (kc + 1) * BK);

        // A fragments: 16-bit A 16x32 layout -> two 16B halves per lane:
        //   elements 0..7  = sA[m][hi*16 + 0..7]     (K = hi*16+e)
        //   elements 8..15 = sA[m][hi*16 + 8 + 0..7] (K = hi*16+8+e)
        v16bf afr[2];
        #pragma unroll
        for (int i = 0; i < 2; ++i) {
            const int m = wm + i * 16 + lm;
            v8bf lo = *(const v8bf*)&sA[cur][m][hi * 16];
            v8bf hh = *(const v8bf*)&sA[cur][m][hi * 16 + 8];
            #pragma unroll
            for (int e = 0; e < 8; ++e) { afr[i][e] = lo[e]; afr[i][8 + e] = hh[e]; }
        }
        // B fragments: 16-bit B 32x16 layout -> lane n = lm, K = hi*16 + e,
        // contiguous 32B from the transposed tile.
        v16bf bfr[4];
        #pragma unroll
        for (int j = 0; j < 4; ++j) {
            const int n = wn + j * 16 + lm;
            bfr[j] = *(const v16bf*)&sB[cur][n][hi * 16];
        }

        #pragma unroll
        for (int i = 0; i < 2; ++i)
            #pragma unroll
            for (int j = 0; j < 4; ++j)
                acc[i][j] = __builtin_amdgcn_wmma_f32_16x16x32_bf16(
                    false, afr[i], false, bfr[j], (short)0, acc[i][j],
                    false, false);

        // Drain this iteration's async fills of buffer cur^1, then make them
        // visible to all waves.
        WAIT_ASYNC();
        __syncthreads();
    }

    // Store: C/D layout — VGPR r: lanes 0-15 hold M=r, lanes 16-31 M=r+8,
    // N = lane&15.
    #pragma unroll
    for (int i = 0; i < 2; ++i)
        #pragma unroll
        for (int j = 0; j < 4; ++j) {
            const int mb = m0 + wm + i * 16 + hi * 8;
            const int nn = n0 + wn + j * 16 + lm;
            #pragma unroll
            for (int r = 0; r < 8; ++r)
                C[(size_t)(mb + r) * N + nn] = acc[i][j][r];
        }
}

// ------------------------------------------------------------------
// out[row][c] = relu( (x+bias - mu) * rsqrt(var+eps) * g + b )  as bf16
// One 256-thread WG (8 waves) per row of 1024.
// ------------------------------------------------------------------
__global__ __launch_bounds__(256)
void bias_ln_relu(const float* __restrict__ Hpre, const float* __restrict__ bias,
                  const float* __restrict__ gamma, const float* __restrict__ beta,
                  __bf16* __restrict__ out) {
    const int row = blockIdx.x;
    const int tid = threadIdx.x;
    const float* x = Hpre + (size_t)row * HID;

    float v[4];
    float s = 0.0f;
    #pragma unroll
    for (int i = 0; i < 4; ++i) {
        const int c = tid + i * 256;
        const float t = x[c] + bias[c];
        v[i] = t; s += t;
    }

    __shared__ float red[8];
    #pragma unroll
    for (int o = 16; o > 0; o >>= 1) s += __shfl_down(s, o, 32);
    if ((tid & 31) == 0) red[tid >> 5] = s;
    __syncthreads();
    float mu = 0.0f;
    #pragma unroll
    for (int w = 0; w < 8; ++w) mu += red[w];
    mu *= (1.0f / (float)HID);

    float s2 = 0.0f;
    #pragma unroll
    for (int i = 0; i < 4; ++i) { const float d = v[i] - mu; s2 += d * d; }
    __syncthreads();                      // before reusing red[]
    #pragma unroll
    for (int o = 16; o > 0; o >>= 1) s2 += __shfl_down(s2, o, 32);
    if ((tid & 31) == 0) red[tid >> 5] = s2;
    __syncthreads();
    float var = 0.0f;
    #pragma unroll
    for (int w = 0; w < 8; ++w) var += red[w];
    var *= (1.0f / (float)HID);

    const float rs = rsqrtf(var + 1e-5f);
    #pragma unroll
    for (int i = 0; i < 4; ++i) {
        const int c = tid + i * 256;
        const float y = (v[i] - mu) * rs * gamma[c] + beta[c];
        out[(size_t)row * HID + c] = (__bf16)fmaxf(y, 0.0f);
    }
}

// ------------------------------------------------------------------
// logit[row] = sigmoid( dot(H2[row], Wout) + bout )   one wave per row
// ------------------------------------------------------------------
__global__ __launch_bounds__(256)
void out_kernel(const __bf16* __restrict__ H2, const float* __restrict__ Wout,
                const float* __restrict__ bout, float* __restrict__ out) {
    const int row  = blockIdx.x * 8 + (threadIdx.x >> 5);
    const int lane = threadIdx.x & 31;
    const __bf16* h = H2 + (size_t)row * HID;

    float s = 0.0f;
    #pragma unroll
    for (int i = 0; i < 32; ++i) {
        const int c = lane + i * 32;
        s = fmaf((float)h[c], Wout[c], s);
    }
    #pragma unroll
    for (int o = 16; o > 0; o >>= 1) s += __shfl_down(s, o, 32);
    if (lane == 0) {
        const float z = s + bout[0];
        out[row] = 1.0f / (1.0f + __expf(-z));
    }
}

// ------------------------------------------------------------------
// Host launcher
// ------------------------------------------------------------------
extern "C" void kernel_launch(void* const* d_in, const int* in_sizes, int n_in,
                              void* d_out, int out_size, void* d_ws, size_t ws_size,
                              hipStream_t stream) {
    // setup_inputs() order
    const float* conts = (const float*)d_in[0];
    const int*   cates = (const int*)  d_in[1];
    /* d_in[2] = combs: unused by the reference */
    const float* emb   = (const float*)d_in[3];
    const float* W1    = (const float*)d_in[4];
    const float* b1    = (const float*)d_in[5];
    const float* ln1g  = (const float*)d_in[6];
    const float* ln1b  = (const float*)d_in[7];
    const float* W2    = (const float*)d_in[8];
    const float* b2    = (const float*)d_in[9];
    const float* ln2g  = (const float*)d_in[10];
    const float* ln2b  = (const float*)d_in[11];
    const float* Wout  = (const float*)d_in[12];
    const float* bout  = (const float*)d_in[13];
    float* out = (float*)d_out;

    // Workspace layout (all offsets 256B-aligned). Total ~216.4 MB.
    char* ws = (char*)d_ws;
    constexpr size_t SZ_X   = (size_t)BATCH * KPAD1 * 2;   // 106,954,752
    constexpr size_t SZ_W1T = (size_t)HID * KPAD1 * 2;     //   6,684,672
    constexpr size_t SZ_W2T = (size_t)HID * HID * 2;       //   2,097,152
    constexpr size_t SZ_HP  = (size_t)BATCH * HID * 4;     //  67,108,864
    __bf16* Xbf  = (__bf16*)(ws);
    __bf16* W1t  = (__bf16*)(ws + SZ_X);
    __bf16* W2t  = (__bf16*)(ws + SZ_X + SZ_W1T);
    float*  Hpre = (float*) (ws + SZ_X + SZ_W1T + SZ_W2T);
    __bf16* H1bf = (__bf16*)(ws + SZ_X + SZ_W1T + SZ_W2T + SZ_HP);
    __bf16* H2bf = Xbf;   // X is dead after GEMM1 -> reuse for H2

    // 1) Weight transpose + bf16 convert (K padded with zeros for WMMA)
    {
        const int total1 = HID * KPAD1;
        transpose_to_bf16<<<(total1 + 255) / 256, 256, 0, stream>>>(W1, W1t,
                                                                    FIRST_RAW, HID, KPAD1);
        const int total2 = HID * HID;
        transpose_to_bf16<<<(total2 + 255) / 256, 256, 0, stream>>>(W2, W2t,
                                                                    HID, HID, HID);
    }

    // 2) Feature build: embeddings + gram pairs -> X (bf16)
    build_X<<<BATCH, 256, 0, stream>>>(conts, cates, emb, Xbf);

    // 3) MLP layer 1: X @ W1 (WMMA bf16 -> f32)
    gemm_bf16_wmma<<<dim3(HID / BN, BATCH / BM), 256, 0, stream>>>(
        Xbf, W1t, Hpre, BATCH, HID, KPAD1);
    bias_ln_relu<<<BATCH, 256, 0, stream>>>(Hpre, b1, ln1g, ln1b, H1bf);

    // 4) MLP layer 2: H1 @ W2 (WMMA bf16 -> f32)
    gemm_bf16_wmma<<<dim3(HID / BN, BATCH / BM), 256, 0, stream>>>(
        H1bf, W2t, Hpre, BATCH, HID, HID);
    bias_ln_relu<<<BATCH, 256, 0, stream>>>(Hpre, b2, ln2g, ln2b, H2bf);

    // 5) Output head + sigmoid
    out_kernel<<<BATCH / 8, 256, 0, stream>>>(H2bf, Wout, bout, out);

    (void)in_sizes; (void)n_in; (void)out_size; (void)ws_size;
}
